// SelfAttention_11544872091926
// MI455X (gfx1250) — matrix operand order repaired and checked by reading, not compiled, not run on hardware
//
#include <hip/hip_runtime.h>
#include <hip/hip_bf16.h>

// ---------------------------------------------------------------------------
// SelfAttention block for MI455X (gfx1250, wave32, v_wmma_f32_16x16x32_f16)
// Flash-attention formulation; all WMMA operand fills are 2x ds_load_b128;
// all global->LDS tile staging uses global_load_async_to_lds_b128 (ASYNCcnt)
// in a double-buffered pipeline.
// Workspace (~36.6 MB):
//   [0, 4KB)              : per-(b,g) mean/rstd
//   [4KB, +160KB)         : wallT f16 (320 x 256)  = [Wq;Wk;Wv] stacked
//   [164KB, +128KB)       : wpt f16 (256 x 256)    = Wproj (already n-major)
//   [512KB, +16MB)        : hn f16 (B,N,C)  -- reused as attn-out later
//   [512KB+16MB, +4MB)    : qk f16 (B,N,64)
//   [512KB+20MB, +16MB)   : vt f16 (B,C,N)  (V stored transposed)
// ---------------------------------------------------------------------------

typedef __attribute__((ext_vector_type(16))) _Float16 v16h;
typedef __attribute__((ext_vector_type(8)))  _Float16 v8h;
typedef __attribute__((ext_vector_type(8)))  float    v8f;

#define Bb   8
#define Cc   256
#define Nn   4096
#define QKVW 320
#define STR  40   // LDS half-stride (80B rows: 16B-aligned, bank-skewed)

__device__ __forceinline__ v16h cat8(v8h lo8, v8h hi8) {
  return __builtin_shufflevector(lo8, hi8, 0,1,2,3,4,5,6,7,8,9,10,11,12,13,14,15);
}
// A-operand fill: row-major row pointer, chunks at k=8*hi and k=16+8*hi
__device__ __forceinline__ v16h fillA(const _Float16* rowp, int hi) {
  const _Float16* p = rowp + 8 * hi;
  return cat8(*(const v8h*)p, *(const v8h*)(p + 16));
}
// B-operand fill: n-major row pointer, chunks at k=8*hi and k=16+8*hi
__device__ __forceinline__ v16h fillB(const _Float16* rowp, int hi) {
  const _Float16* p = rowp + 8 * hi;
  return cat8(*(const v8h*)p, *(const v8h*)(p + 16));
}

// Generic pointer to LDS byte offset (low 32 bits of the aperture address)
__device__ __forceinline__ unsigned lds_off(const void* p) {
  return (unsigned)(unsigned long long)p;
}
// Async global->LDS 16B copy, GVS mode (SGPR base + 32-bit VGPR byte offset)
__device__ __forceinline__ void async_b128(unsigned lds, unsigned goff,
                                           const void* base) {
  asm volatile("global_load_async_to_lds_b128 %0, %1, %2"
               :: "v"(lds), "v"(goff), "s"((unsigned long long)base)
               : "memory");
}
template<int N>
__device__ __forceinline__ void wait_asynccnt() {
  asm volatile("s_wait_asynccnt %0" :: "n"(N) : "memory");
}

// ------------------------- GroupNorm statistics ----------------------------
__global__ void gn_stats(const float* __restrict__ x, float* __restrict__ stats) {
  int grp = blockIdx.x;                       // b*32+g ; group channels contiguous
  const float* px = x + (size_t)grp * 8 * Nn;
  float s = 0.f, sq = 0.f;
  for (int i = threadIdx.x; i < 8 * Nn; i += blockDim.x) {
    float v = px[i]; s += v; sq += v * v;
  }
  for (int m = 16; m >= 1; m >>= 1) { s += __shfl_xor(s, m, 32); sq += __shfl_xor(sq, m, 32); }
  __shared__ float ls[8], lq[8];
  int wave = threadIdx.x >> 5, lane = threadIdx.x & 31;
  if (lane == 0) { ls[wave] = s; lq[wave] = sq; }
  __syncthreads();
  if (threadIdx.x == 0) {
    float S = 0.f, Q = 0.f;
    for (int i = 0; i < 8; i++) { S += ls[i]; Q += lq[i]; }
    float mean = S / 32768.f;
    float var  = Q / 32768.f - mean * mean;
    stats[grp * 2]     = mean;
    stats[grp * 2 + 1] = rsqrtf(var + 1e-5f);
  }
}

// --------------- normalize + scale/shift + transpose to (B,N,C) f16 --------
__global__ void gn_apply_t(const float* __restrict__ x,
                           const float* __restrict__ nw,
                           const float* __restrict__ nb,
                           const float* __restrict__ stats,
                           _Float16* __restrict__ hn) {
  int n0 = blockIdx.x * 32, c0 = blockIdx.y * 32, b = blockIdx.z;
  __shared__ float tile[32][33];
  int tx = threadIdx.x, ty = threadIdx.y;
  for (int i = 0; i < 4; i++) {
    int c = c0 + ty + i * 8;
    int g = c >> 3;
    float mean = stats[(b * 32 + g) * 2];
    float rstd = stats[(b * 32 + g) * 2 + 1];
    float v = x[((size_t)(b * Cc + c)) * Nn + n0 + tx];
    tile[ty + i * 8][tx] = (v - mean) * rstd * nw[c] + nb[c];
  }
  __syncthreads();
  for (int i = 0; i < 4; i++) {
    int n = n0 + ty + i * 8;
    int c = c0 + tx;
    hn[((size_t)b * Nn + n) * Cc + c] = (_Float16)tile[tx][ty + i * 8];
  }
}

// -------- pack weights: wallT = [Wq;Wk;Wv] (320x256), wpt = Wproj (256x256) -
__global__ void pack_weights(const float* __restrict__ Wq, const float* __restrict__ Wk,
                             const float* __restrict__ Wv, const float* __restrict__ Wp,
                             _Float16* __restrict__ wallT, _Float16* __restrict__ wpt) {
  int idx = blockIdx.x * blockDim.x + threadIdx.x;
  if (idx < QKVW * Cc) {
    int j = idx >> 8;
    float v = (j < 32) ? Wq[idx] : (j < 64) ? Wk[idx - 32 * Cc] : Wv[idx - 64 * Cc];
    wallT[idx] = (_Float16)v;
  }
  int idx2 = idx - QKVW * Cc;
  if (idx2 >= 0 && idx2 < Cc * Cc) wpt[idx2] = (_Float16)Wp[idx2];
}

// -------- QKV GEMM: hn(32768x256) x W^T ; Q|K row-major, V transposed -------
__global__ void gemm_qkv(const _Float16* __restrict__ hn,
                         const _Float16* __restrict__ wallT,
                         _Float16* __restrict__ qk,
                         _Float16* __restrict__ vt) {
  // grid (5, 512), block 128 (4 waves). 64x64 block tile, K-step 32.
  __shared__ __attribute__((aligned(16))) _Float16 a_lds[2][64 * STR];
  __shared__ __attribute__((aligned(16))) _Float16 b_lds[2][64 * STR];  // n-major
  int tn = blockIdx.x, tm = blockIdx.y;
  int tid = threadIdx.x, lane = tid & 31, wave = tid >> 5;
  int hi = lane >> 4, ln = lane & 15;

  auto issue = [&](int kb, int buf) {       // 4 async b128 per thread per tile
    for (int t = 0; t < 2; t++) {
      int v = tid + t * 128, row = v >> 2, seg = v & 3;
      async_b128(lds_off(&a_lds[buf][row * STR + seg * 8]),
                 (unsigned)(((unsigned)(tm * 64 + row) * Cc + kb * 32 + seg * 8) * 2), hn);
      async_b128(lds_off(&b_lds[buf][row * STR + seg * 8]),
                 (unsigned)(((unsigned)(tn * 64 + row) * Cc + kb * 32 + seg * 8) * 2), wallT);
    }
  };
  issue(0, 0); issue(1, 1);

  v8f acc[4] = {};
  for (int kb = 0; kb < 8; kb++) {
    if (kb < 7) wait_asynccnt<4>(); else wait_asynccnt<0>();
    __syncthreads();
    int buf = kb & 1;
    v16h a = fillA(a_lds[buf] + (wave * 16 + ln) * STR, hi);
    for (int t = 0; t < 4; t++) {
      v16h bop = fillB(b_lds[buf] + (t * 16 + ln) * STR, hi);
      acc[t] = __builtin_amdgcn_wmma_f32_16x16x32_f16(false, a, false, bop,
                                                      (short)0, acc[t], false, false);
    }
    __syncthreads();
    if (kb + 2 < 8) issue(kb + 2, buf);
  }

  int bb = (tm * 64) >> 12;                 // batch (tile never straddles)
  int nbase = ((tm * 64) & 4095) + wave * 16 + 8 * hi;
  if (tn == 0) {                            // Q|K -> row-major (B,N,64)
    for (int t = 0; t < 4; t++)
      for (int vg = 0; vg < 8; vg++) {
        size_t m = (size_t)tm * 64 + wave * 16 + vg + 8 * hi;
        qk[m * 64 + t * 16 + ln] = (_Float16)acc[t][vg];
      }
  } else {                                  // V -> transposed (B,C,N), 128b stores
    for (int t = 0; t < 4; t++) {
      v8h hv;
      for (int vg = 0; vg < 8; vg++) hv[vg] = (_Float16)acc[t][vg];
      int ch = (tn - 1) * 64 + t * 16 + ln;
      *(v8h*)(vt + ((size_t)(bb * Cc + ch)) * Nn + nbase) = hv;
    }
  }
}

// ------------------------------ flash attention -----------------------------
__global__ void flash_attn(const _Float16* __restrict__ qk,
                           const _Float16* __restrict__ vt,
                           _Float16* __restrict__ attnout) {
  // grid (N/64, B), block 128 (4 waves x 16 Q-rows). Bc = 32, double-buffered.
  __shared__ __attribute__((aligned(16))) _Float16 kt_lds[2][32 * STR];   // n-major
  __shared__ __attribute__((aligned(16))) _Float16 v_lds[2][256 * STR];   // ch-major
  __shared__ __attribute__((aligned(16))) _Float16 p_lds[4][16 * STR];    // row-major
  __shared__ __attribute__((aligned(16))) _Float16 ob_lds[4][16 * 264];
  int tq = blockIdx.x, b = blockIdx.y;
  int tid = threadIdx.x, lane = tid & 31, wave = tid >> 5;
  int hi = lane >> 4, ln = lane & 15;
  size_t rowbase = (size_t)b * Nn;

  int qrow = tq * 64 + wave * 16 + ln;
  v16h qa = fillA(qk + (rowbase + qrow) * 64, hi);   // Q cols 0..31, 2x b128

  // 5 async b128 per thread per tile (1 K-seg + 4 V-segs), in-order completion
  auto issue_tile = [&](int j, int buf) {
    int krow = tid >> 2, kseg = tid & 3;
    async_b128(lds_off(&kt_lds[buf][krow * STR + kseg * 8]),
               (unsigned)((((unsigned)b * Nn + j * 32 + krow) * 64 + 32 + kseg * 8) * 2), qk);
    for (int t = 0; t < 2; t++) {
      int v = tid + t * 128, row = v >> 1, seg = v & 1;
      unsigned l0 = lds_off(&v_lds[buf][row * STR + seg * 16]);
      unsigned g0 = (unsigned)((((unsigned)(b * Cc + row)) * Nn + j * 32 + seg * 16) * 2);
      async_b128(l0,      g0,      vt);
      async_b128(l0 + 16, g0 + 32, vt);   // +8 halves
    }
  };
  issue_tile(0, 0); issue_tile(1, 1);

  v8f acc[16] = {};
  float mrow[8], lrow[8];
  for (int i = 0; i < 8; i++) { mrow[i] = -3.0e38f; lrow[i] = 0.f; }
  const float scale = 0.17677669529663687f;  // 1/sqrt(32)

  for (int j = 0; j < 128; j++) {
    if (j < 127) wait_asynccnt<5>(); else wait_asynccnt<0>();
    __syncthreads();   // whole tile j visible to all waves
    int buf = j & 1;

    // S = Q*K^T : two 16x16 tiles, single WMMA each (K-dim = d = 32)
    v16h b0 = fillB(kt_lds[buf] + ln * STR, hi);
    v16h b1 = fillB(kt_lds[buf] + (16 + ln) * STR, hi);
    v8f s0 = {}, s1 = {};
    s0 = __builtin_amdgcn_wmma_f32_16x16x32_f16(false, qa, false, b0, (short)0, s0, false, false);
    s1 = __builtin_amdgcn_wmma_f32_16x16x32_f16(false, qa, false, b1, (short)0, s1, false, false);

    // Online softmax: row m = vg + 8*hi lives across the 16 lanes of a half.
    for (int vg = 0; vg < 8; vg++) {
      float a0 = s0[vg] * scale, a1 = s1[vg] * scale;
      float rmax = fmaxf(a0, a1);
      for (int m = 8; m >= 1; m >>= 1) rmax = fmaxf(rmax, __shfl_xor(rmax, m, 32));
      float mnew = fmaxf(mrow[vg], rmax);
      float corr = __expf(mrow[vg] - mnew);
      float p0 = __expf(a0 - mnew);
      float p1 = __expf(a1 - mnew);
      float ps = p0 + p1;
      for (int m = 8; m >= 1; m >>= 1) ps += __shfl_xor(ps, m, 32);
      lrow[vg] = lrow[vg] * corr + ps;
      mrow[vg] = mnew;
      for (int t = 0; t < 16; t++) acc[t][vg] *= corr;
      p_lds[wave][(vg + 8 * hi) * STR + ln]      = (_Float16)p0;
      p_lds[wave][(vg + 8 * hi) * STR + 16 + ln] = (_Float16)p1;
    }
    asm volatile("s_wait_dscnt 0" ::: "memory");  // same-wave DS ordering (in-order)

    v16h pa = fillA(p_lds[wave] + ln * STR, hi);  // P in A-layout, 2x b128
    for (int t = 0; t < 16; t++) {
      v16h bv = fillB(v_lds[buf] + (t * 16 + ln) * STR, hi);
      acc[t] = __builtin_amdgcn_wmma_f32_16x16x32_f16(false, pa, false, bv,
                                                      (short)0, acc[t], false, false);
    }
    __syncthreads();   // everyone done reading buf before it is refilled
    if (j + 2 < 128) issue_tile(j + 2, buf);
  }

  // normalize, bounce through LDS, store coalesced 128b rows
  for (int t = 0; t < 16; t++)
    for (int vg = 0; vg < 8; vg++)
      ob_lds[wave][(vg + 8 * hi) * 264 + t * 16 + ln] = (_Float16)(acc[t][vg] / lrow[vg]);
  asm volatile("s_wait_dscnt 0" ::: "memory");
  int mbase = tq * 64 + wave * 16;
  for (int i = 0; i < 16; i++) {
    v8h row = *(const v8h*)(ob_lds[wave] + i * 264 + lane * 8);
    *(v8h*)(attnout + (rowbase + mbase + i) * Cc + lane * 8) = row;
  }
}

// ---------------- out projection + residual: x + gamma * (out @ Wproj^T) ----
__global__ void gemm_proj(const _Float16* __restrict__ ao,
                          const _Float16* __restrict__ wpt,
                          const float* __restrict__ x,
                          const float* __restrict__ gamma,
                          float* __restrict__ out) {
  __shared__ __attribute__((aligned(16))) _Float16 a_lds[2][64 * STR];
  __shared__ __attribute__((aligned(16))) _Float16 b_lds[2][64 * STR];  // n(o)-major
  int tn = blockIdx.x, tm = blockIdx.y;     // grid (4, 512)
  int tid = threadIdx.x, lane = tid & 31, wave = tid >> 5;
  int hi = lane >> 4, ln = lane & 15;

  auto issue = [&](int kb, int buf) {
    for (int t = 0; t < 2; t++) {
      int v = tid + t * 128, row = v >> 2, seg = v & 3;
      async_b128(lds_off(&a_lds[buf][row * STR + seg * 8]),
                 (unsigned)(((unsigned)(tm * 64 + row) * Cc + kb * 32 + seg * 8) * 2), ao);
      async_b128(lds_off(&b_lds[buf][row * STR + seg * 8]),
                 (unsigned)(((unsigned)(tn * 64 + row) * Cc + kb * 32 + seg * 8) * 2), wpt);
    }
  };
  issue(0, 0); issue(1, 1);

  v8f acc[4] = {};
  for (int kb = 0; kb < 8; kb++) {
    if (kb < 7) wait_asynccnt<4>(); else wait_asynccnt<0>();
    __syncthreads();
    int buf = kb & 1;
    v16h a = fillA(a_lds[buf] + (wave * 16 + ln) * STR, hi);
    for (int t = 0; t < 4; t++) {
      v16h bop = fillB(b_lds[buf] + (t * 16 + ln) * STR, hi);
      acc[t] = __builtin_amdgcn_wmma_f32_16x16x32_f16(false, a, false, bop,
                                                      (short)0, acc[t], false, false);
    }
    __syncthreads();
    if (kb + 2 < 8) issue(kb + 2, buf);
  }

  float g = gamma[0];
  int bb = (tm * 64) >> 12;
  int nbase = ((tm * 64) & 4095) + wave * 16 + 8 * hi;   // 8 consecutive n per lane
  for (int t = 0; t < 4; t++) {
    int o = tn * 64 + t * 16 + ln;
    size_t idx = ((size_t)(bb * Cc + o)) * Nn + nbase;
    float4 x0 = *(const float4*)(x + idx);
    float4 x1 = *(const float4*)(x + idx + 4);
    float4 o0, o1;
    o0.x = x0.x + g * acc[t][0]; o0.y = x0.y + g * acc[t][1];
    o0.z = x0.z + g * acc[t][2]; o0.w = x0.w + g * acc[t][3];
    o1.x = x1.x + g * acc[t][4]; o1.y = x1.y + g * acc[t][5];
    o1.z = x1.z + g * acc[t][6]; o1.w = x1.w + g * acc[t][7];
    *(float4*)(out + idx)     = o0;
    *(float4*)(out + idx + 4) = o1;
  }
}

// ---------------------------------------------------------------------------
extern "C" void kernel_launch(void* const* d_in, const int* in_sizes, int n_in,
                              void* d_out, int out_size, void* d_ws, size_t ws_size,
                              hipStream_t stream) {
  const float* x      = (const float*)d_in[0];
  const float* norm_w = (const float*)d_in[1];
  const float* norm_b = (const float*)d_in[2];
  const float* Wq     = (const float*)d_in[3];
  const float* Wk     = (const float*)d_in[4];
  const float* Wv     = (const float*)d_in[5];
  const float* Wproj  = (const float*)d_in[6];
  const float* gamma  = (const float*)d_in[7];
  float* out = (float*)d_out;

  char* ws = (char*)d_ws;
  float*    stats = (float*)ws;
  _Float16* wallT = (_Float16*)(ws + 4096);
  _Float16* wpt   = (_Float16*)(ws + 4096 + 163840);
  _Float16* hn    = (_Float16*)(ws + (1 << 19));                     // 16MB
  _Float16* qk    = (_Float16*)(ws + (1 << 19) + (16 << 20));        // 4MB
  _Float16* vt    = (_Float16*)(ws + (1 << 19) + (20 << 20));        // 16MB
  _Float16* attnout = hn;  // hn dead after gemm_qkv -> reuse

  gn_stats<<<256, 256, 0, stream>>>(x, stats);
  gn_apply_t<<<dim3(Nn / 32, Cc / 32, Bb), dim3(32, 8), 0, stream>>>(x, norm_w, norm_b, stats, hn);
  pack_weights<<<576, 256, 0, stream>>>(Wq, Wk, Wv, Wproj, wallT, wpt);
  gemm_qkv<<<dim3(5, 512), 128, 0, stream>>>(hn, wallT, qk, vt);
  flash_attn<<<dim3(Nn / 64, Bb), 128, 0, stream>>>(qk, vt, attnout);
  gemm_proj<<<dim3(4, 512), 128, 0, stream>>>(attnout, wpt, x, gamma, out);
}
